// GraphDecoder_6287832121921
// MI455X (gfx1250) — compile-verified, matrix operand order checked by reading, not run on hardware
//
#include <hip/hip_runtime.h>

typedef _Float16 half_t;
typedef __attribute__((ext_vector_type(16))) _Float16 v16h;
typedef __attribute__((ext_vector_type(8)))  _Float16 v8h;
typedef __attribute__((ext_vector_type(8)))  float    v8f;
typedef __attribute__((ext_vector_type(4)))  int      v4i;

#define SLOPE 0.2f

#if __has_builtin(__builtin_amdgcn_global_load_async_to_lds_b128) && \
    __has_builtin(__builtin_amdgcn_s_wait_asynccnt)
#define HAVE_ASYNC_LDS 1
#else
#define HAVE_ASYNC_LDS 0
#endif

static __device__ __forceinline__ void atomicMaxF(float* addr, float val) {
    int* a = (int*)addr;
    int old = __hip_atomic_load(a, __ATOMIC_RELAXED, __HIP_MEMORY_SCOPE_AGENT);
    while (__int_as_float(old) < val) {
        int assumed = old;
        old = atomicCAS(a, assumed, __float_as_int(val));
        if (old == assumed) break;
    }
}

// ---------- layer-1 structural shortcut ----------
// gfeat[g,f] = relu(z[g,:]@W0[:,f] + b0[f])    (G x 64)
__global__ void k_gfeat(const float* __restrict__ z, const float* __restrict__ W0,
                        const float* __restrict__ b0, float* __restrict__ gf, int G) {
    int gid = blockIdx.x * blockDim.x + threadIdx.x;
    if (gid >= G * 64) return;
    int g = gid >> 6, f = gid & 63;
    float acc = b0[f];
    const float* zp = z + (size_t)g * 256;
    #pragma unroll 8
    for (int k = 0; k < 256; ++k) acc += zp[k] * W0[(size_t)k * 64 + f];
    gf[gid] = acc > 0.f ? acc : 0.f;
}

// gh[g,j] = gfeat[g,:]@W1[0:64, j]   (G x 256)
__global__ void k_gh(const float* __restrict__ gf, const float* __restrict__ W1,
                     float* __restrict__ gh, int G) {
    int gid = blockIdx.x * blockDim.x + threadIdx.x;
    if (gid >= G * 256) return;
    int g = gid >> 8, j = gid & 255;
    float acc = 0.f;
    const float* gp = gf + (size_t)g * 64;
    #pragma unroll 8
    for (int k = 0; k < 64; ++k) acc += gp[k] * W1[(size_t)k * 256 + j];
    gh[gid] = acc;
}

// h1[n,j] = gh[batch[n],j] + W1[64+order(n), j]
__global__ void k_h1(const float* __restrict__ gh, const float* __restrict__ W1,
                     const int* __restrict__ batch, float* __restrict__ h, int n, int npg) {
    int gid = blockIdx.x * blockDim.x + threadIdx.x;
    if (gid >= n * 256) return;
    int node = gid >> 8, j = gid & 255;
    int g = batch[node];
    int o = node % npg;
    h[gid] = gh[(size_t)g * 256 + j] + W1[(size_t)(64 + o) * 256 + j];
}

// ---------- generic pieces ----------
// f32 [K,NC] -> f16 transposed [NC,K]
__global__ void k_w_half_t(const float* __restrict__ W, half_t* __restrict__ Wt,
                           int K, int NC) {
    int gid = blockIdx.x * blockDim.x + threadIdx.x;
    if (gid >= K * NC) return;
    int k = gid / NC, j = gid % NC;
    Wt[(size_t)j * K + k] = (half_t)W[gid];
}

// WMMA GEMM: C[nrows,ncols] = A16[nrows,256] * Bt16[ncols,256]^T, fp32 accum.
// Block = 8 waves = 128x64 macro-tile. 64-col weight slice staged in LDS
// (async-to-LDS when available), each wave: 1 row tile x 4 col tiles,
// A fragment reused across 4 WMMAs per k-step -> 32 v_wmma per wave.
#define BSTRIDE 264   // 256 + 8 halves pad: 528B row stride -> conflict-free frag reads
__global__ __launch_bounds__(256) void k_gemm_wmma(const half_t* __restrict__ A,
                                                   const half_t* __restrict__ Bt,
                                                   float* __restrict__ C,
                                                   int nrows, int ncols) {
    const int K = 256;
    __shared__ half_t Bs[64 * BSTRIDE];          // 33 KB weight slice
    const int wave = threadIdx.x >> 5;
    const int lane = threadIdx.x & 31;
    const int cgBase = blockIdx.y * 64;          // first output column of block
    const int row0   = blockIdx.x * 128 + wave * 16;

    // ---- stage Bt[cgBase..cgBase+64) x K into LDS (16B per lane per op) ----
    {
        const half_t* src = Bt + (size_t)cgBase * K;
        #pragma unroll
        for (int it = 0; it < 8; ++it) {
            int i = it * 256 + threadIdx.x;      // 2048 chunks of 8 halves
            int r = i >> 5, c = i & 31;
            const half_t* gp = src + (size_t)r * K + c * 8;
            half_t* lp = Bs + (size_t)r * BSTRIDE + c * 8;
#if HAVE_ASYNC_LDS
            __builtin_amdgcn_global_load_async_to_lds_b128((v4i*)gp, (v4i*)lp, 0, 0);
#else
            *(v8h*)lp = *(const v8h*)gp;
#endif
        }
    }
#if HAVE_ASYNC_LDS
    __builtin_amdgcn_s_wait_asynccnt(0);
#endif
    __syncthreads();

    const int m = lane & 15;      // A row / B col within 16x16 tile
    const int g = lane >> 4;      // K half-group select
    const half_t* arow = A + (size_t)(row0 + m) * K;

    v8f acc[4] = {};
    #pragma unroll
    for (int kk = 0; kk < K; kk += 32) {
        v8h a0 = *(const v8h*)(arow + kk +      g * 8);
        v8h a1 = *(const v8h*)(arow + kk + 16 + g * 8);
        v16h av;
        #pragma unroll
        for (int i = 0; i < 8; ++i) { av[i] = a0[i]; av[i + 8] = a1[i]; }
        #pragma unroll
        for (int ct = 0; ct < 4; ++ct) {
            const half_t* brow = Bs + (size_t)(ct * 16 + m) * BSTRIDE;
            v8h b0 = *(const v8h*)(brow + kk +      g * 8);   // ds_load_b128
            v8h b1 = *(const v8h*)(brow + kk + 16 + g * 8);
            v16h bv;
            #pragma unroll
            for (int i = 0; i < 8; ++i) { bv[i] = b0[i]; bv[i + 8] = b1[i]; }
            acc[ct] = __builtin_amdgcn_wmma_f32_16x16x32_f16(
                false, av, false, bv, (short)0, acc[ct], false, false);
        }
    }
    // C/D layout: VGPR r -> M = r + 8*(lane>=16), N = lane&15
    const int rbase = row0 + g * 8;
    #pragma unroll
    for (int ct = 0; ct < 4; ++ct) {
        int col = cgBase + ct * 16 + m;
        #pragma unroll
        for (int r = 0; r < 8; ++r)
            C[(size_t)(rbase + r) * ncols + col] = acc[ct][r];
    }
}

// a_s[n,hd] = h[n,hd,:].att_s[hd,:] ; same for a_d
__global__ void k_att(const float* __restrict__ h, const float* __restrict__ as_,
                      const float* __restrict__ ad_, float* __restrict__ a_s,
                      float* __restrict__ a_d, int n) {
    int gid = blockIdx.x * blockDim.x + threadIdx.x;
    if (gid >= n * 4) return;
    int node = gid >> 2, hd = gid & 3;
    const float* hp = h + (size_t)node * 256 + hd * 64;
    const float* sp = as_ + hd * 64;
    const float* dp = ad_ + hd * 64;
    float ss = 0.f, dd = 0.f;
    #pragma unroll 8
    for (int f = 0; f < 64; ++f) { ss += hp[f] * sp[f]; dd += hp[f] * dp[f]; }
    a_s[gid] = ss;
    a_d[gid] = dd;
}

__global__ void k_seg_init(float* __restrict__ agg, float* __restrict__ m,
                           float* __restrict__ s, int n) {
    int gid = blockIdx.x * blockDim.x + threadIdx.x;
    if (gid < n * 256) agg[gid] = 0.f;
    if (gid < n * 4) { m[gid] = -3.0e38f; s[gid] = 0.f; }
}

// pass 1: logits + segment max over dst
__global__ void k_edge_max(const int* __restrict__ src, const int* __restrict__ dst,
                           const float* __restrict__ a_s, const float* __restrict__ a_d,
                           float* __restrict__ ex, float* __restrict__ m, int E, int Etot) {
    int gid = blockIdx.x * blockDim.x + threadIdx.x;
    if (gid >= Etot * 4) return;
    int e = gid >> 2, hd = gid & 3;
    int sn = (e < E) ? src[e] : (e - E);
    int dn = (e < E) ? dst[e] : (e - E);
    float v = a_s[(size_t)sn * 4 + hd] + a_d[(size_t)dn * 4 + hd];
    v = v > 0.f ? v : SLOPE * v;
    ex[gid] = v;
    atomicMaxF(&m[(size_t)dn * 4 + hd], v);
}

// pass 2: exp(e - max) + segment sum
__global__ void k_edge_exp(const int* __restrict__ src, const int* __restrict__ dst,
                           float* __restrict__ ex, const float* __restrict__ m,
                           float* __restrict__ s, int E, int Etot) {
    int gid = blockIdx.x * blockDim.x + threadIdx.x;
    if (gid >= Etot * 4) return;
    int e = gid >> 2, hd = gid & 3;
    int dn = (e < E) ? dst[e] : (e - E);
    float v = __expf(ex[gid] - m[(size_t)dn * 4 + hd]);
    ex[gid] = v;
    atomicAdd(&s[(size_t)dn * 4 + hd], v);
}

// pass 3: agg[dst] += alpha * h[src], 64 threads/edge, float4 granularity
__global__ void k_edge_scat(const int* __restrict__ src, const int* __restrict__ dst,
                            const float* __restrict__ ex, const float* __restrict__ s,
                            const float* __restrict__ h, float* __restrict__ agg,
                            int E, int Etot) {
    int gid = blockIdx.x * blockDim.x + threadIdx.x;
    if (gid >= Etot * 64) return;
    int e = gid >> 6, t = gid & 63;
    int sn = (e < E) ? src[e] : (e - E);
    int dn = (e < E) ? dst[e] : (e - E);
    int hd = t >> 4;
    float alpha = ex[(size_t)e * 4 + hd] / (s[(size_t)dn * 4 + hd] + 1e-16f);
    const float4 hv = *(const float4*)(h + (size_t)sn * 256 + t * 4);
    float* ap = agg + (size_t)dn * 256 + t * 4;
    atomicAdd(ap + 0, alpha * hv.x);
    atomicAdd(ap + 1, alpha * hv.y);
    atomicAdd(ap + 2, alpha * hv.z);
    atomicAdd(ap + 3, alpha * hv.w);
}

// x16 = (half)relu(agg + bias)   -> feeds next WMMA GEMM
__global__ void k_relu_bias_half(const float* __restrict__ agg, const float* __restrict__ bias,
                                 half_t* __restrict__ x16, int n) {
    int gid = blockIdx.x * blockDim.x + threadIdx.x;
    if (gid >= n * 256) return;
    float v = agg[gid] + bias[gid & 255];
    x16[gid] = (half_t)(v > 0.f ? v : 0.f);
}

// out[n,c] = relu(yraw[n,:]+bg) @ Wf[:,c] + bf[c]
__global__ void k_final(const float* __restrict__ yraw, const float* __restrict__ bg,
                        const float* __restrict__ Wf, const float* __restrict__ bf,
                        float* __restrict__ out, int n) {
    int gid = blockIdx.x * blockDim.x + threadIdx.x;
    if (gid >= n * 5) return;
    int node = gid / 5, c = gid % 5;
    const float* yp = yraw + (size_t)node * 64;
    float acc = bf[c];
    #pragma unroll 8
    for (int f = 0; f < 64; ++f) {
        float v = yp[f] + bg[f];
        v = v > 0.f ? v : 0.f;
        acc += v * Wf[(size_t)f * 5 + c];
    }
    out[gid] = acc;
}

static inline int cdiv(long a, int b) { return (int)((a + b - 1) / b); }

extern "C" void kernel_launch(void* const* d_in, const int* in_sizes, int n_in,
                              void* d_out, int out_size, void* d_ws, size_t ws_size,
                              hipStream_t stream) {
    const float* z    = (const float*)d_in[0];
    const int*   eix  = (const int*)  d_in[1];
    const int*   batch= (const int*)  d_in[2];
    const float* W0   = (const float*)d_in[4];
    const float* b0   = (const float*)d_in[5];
    const float* W1   = (const float*)d_in[6];
    const float* as1  = (const float*)d_in[7];
    const float* ad1  = (const float*)d_in[8];
    const float* bb1  = (const float*)d_in[9];
    const float* W2   = (const float*)d_in[10];
    const float* as2  = (const float*)d_in[11];
    const float* ad2  = (const float*)d_in[12];
    const float* bb2  = (const float*)d_in[13];
    const float* W3   = (const float*)d_in[14];
    const float* as3  = (const float*)d_in[15];
    const float* ad3  = (const float*)d_in[16];
    const float* bb3  = (const float*)d_in[17];
    const float* Wg   = (const float*)d_in[18];
    const float* bg   = (const float*)d_in[19];
    const float* Wf   = (const float*)d_in[20];
    const float* bf   = (const float*)d_in[21];

    const int G    = in_sizes[0] / 256;
    const int N    = in_sizes[2];
    const int NPG  = N / G;
    const int E    = in_sizes[1] / 2;
    const int Etot = E + N;
    const int* srcp = eix;
    const int* dstp = eix + E;

    // ---- workspace layout ----
    char* ws = (char*)d_ws;
    size_t off = 0;
    auto alloc = [&](size_t bytes) {
        char* p = ws + off;
        off += (bytes + 255) & ~(size_t)255;
        return p;
    };
    half_t* x16  = (half_t*)alloc((size_t)N * 256 * sizeof(half_t));
    float*  h    = (float*) alloc((size_t)N * 256 * sizeof(float));
    float*  agg  = (float*) alloc((size_t)N * 256 * sizeof(float));
    float*  a_s  = (float*) alloc((size_t)N * 4 * sizeof(float));
    float*  a_d  = (float*) alloc((size_t)N * 4 * sizeof(float));
    float*  mseg = (float*) alloc((size_t)N * 4 * sizeof(float));
    float*  sseg = (float*) alloc((size_t)N * 4 * sizeof(float));
    float*  ex   = (float*) alloc((size_t)Etot * 4 * sizeof(float));
    half_t* w16t = (half_t*)alloc((size_t)256 * 256 * sizeof(half_t));
    float*  gfeat= (float*) alloc((size_t)G * 64 * sizeof(float));
    float*  gh   = (float*) alloc((size_t)G * 256 * sizeof(float));
    float*  yraw = (float*) alloc((size_t)N * 64 * sizeof(float));
    (void)ws_size;

    const int B = 256;

    // attention + segment-softmax + scatter + relu, shared by all 3 layers
    auto att_edges = [&](const float* att_s, const float* att_d, const float* bias) {
        k_att<<<cdiv((long)N * 4, B), B, 0, stream>>>(h, att_s, att_d, a_s, a_d, N);
        k_seg_init<<<cdiv((long)N * 256, B), B, 0, stream>>>(agg, mseg, sseg, N);
        k_edge_max<<<cdiv((long)Etot * 4, B), B, 0, stream>>>(srcp, dstp, a_s, a_d, ex, mseg, E, Etot);
        k_edge_exp<<<cdiv((long)Etot * 4, B), B, 0, stream>>>(srcp, dstp, ex, mseg, sseg, E, Etot);
        k_edge_scat<<<cdiv((long)Etot * 64, B), B, 0, stream>>>(srcp, dstp, ex, sseg, h, agg, E, Etot);
        k_relu_bias_half<<<cdiv((long)N * 256, B), B, 0, stream>>>(agg, bias, x16, N);
    };

    // ---- layer 1 (GEMM collapsed via concat structure) ----
    k_gfeat<<<cdiv((long)G * 64, B), B, 0, stream>>>(z, W0, b0, gfeat, G);
    k_gh<<<cdiv((long)G * 256, B), B, 0, stream>>>(gfeat, W1, gh, G);
    k_h1<<<cdiv((long)N * 256, B), B, 0, stream>>>(gh, W1, batch, h, N, NPG);
    att_edges(as1, ad1, bb1);

    // ---- layers 2 & 3 (WMMA GEMMs, 128x64 macro-tiles, LDS-staged weights) ----
    const float* Ws[2]  = { W2, W3 };
    const float* ass[2] = { as2, as3 };
    const float* ads[2] = { ad2, ad3 };
    const float* bbs[2] = { bb2, bb3 };
    for (int l = 0; l < 2; ++l) {
        k_w_half_t<<<cdiv(256L * 256, B), B, 0, stream>>>(Ws[l], w16t, 256, 256);
        dim3 grid(N / 128, 256 / 64);
        k_gemm_wmma<<<grid, B, 0, stream>>>(x16, w16t, h, N, 256);
        att_edges(ass[l], ads[l], bbs[l]);
    }

    // ---- geo head: relu(x@Wg+bg) @ Wf + bf ----
    k_w_half_t<<<cdiv(256L * 64, B), B, 0, stream>>>(Wg, w16t, 256, 64);
    dim3 grid_g(N / 128, 64 / 64);
    k_gemm_wmma<<<grid_g, B, 0, stream>>>(x16, w16t, yraw, N, 64);
    k_final<<<cdiv((long)N * 5, B), B, 0, stream>>>(yraw, bg, Wf, bf, (float*)d_out, N);
}
